// Gemma4TextExperts_83665962926755
// MI455X (gfx1250) — matrix operand order repaired and checked by reading, not compiled, not run on hardware
//
#include <hip/hip_runtime.h>

// ---------------- types ----------------
typedef __bf16 bf16;
typedef __attribute__((ext_vector_type(16))) __bf16 v16bf;
typedef __attribute__((ext_vector_type(8)))  __bf16 v8bf;
typedef __attribute__((ext_vector_type(4)))  __bf16 v4bf;
typedef __attribute__((ext_vector_type(8)))  float  v8f;
typedef __attribute__((ext_vector_type(4)))  unsigned int v4u;
typedef __attribute__((ext_vector_type(8)))  int v8i;
typedef __attribute__((ext_vector_type(4)))  int v4i;

// ---------------- problem constants ----------------
#define T_TOKENS  2048
#define HIDDEN    1024
#define INTER     2048
#define NEXPERT   8
#define TOPK      2
#define MTILE     32                    // tokens per block (2 x 16-row WMMA tiles)
#define CHUNK     512                   // INTER columns per LDS pass
#define NCHUNK    (INTER / CHUNK)       // 4
#define MAX_ROWS  (T_TOKENS * TOPK)     // 4096 worst-case rows per expert
#define MAX_TILES (MAX_ROWS / MTILE)    // 128
#define XS_STRIDE (HIDDEN + 8)          // pad -> conflict-free b128 LDS reads
#define HS_STRIDE (CHUNK + 8)

#define XS_BYTES    (MTILE * XS_STRIDE * 2)          // 66048
#define STAGE_BYTES (16 * HIDDEN * 4)                // 65536: fp32 TDM landing pad (>= HS bytes)
#define LDS_BYTES   (XS_BYTES + STAGE_BYTES + MTILE * 4 + MTILE * 4)

// ---------------- helpers ----------------
// gelu_tanh(x) = x * sigmoid(2*sqrt(2/pi)*(x + 0.044715*x^3)); branchless,
// raw v_exp_f32 / v_rcp_f32 (TRANS ops co-execute with XDL WMMA).
static __device__ __forceinline__ float gelu_tanh(float x) {
    float z = x + 0.044715f * x * x * x;
    float e = __builtin_amdgcn_exp2f(-2.3022081836f * z);   // -2*sqrt(2/pi)*log2(e)
    return x * __builtin_amdgcn_rcpf(1.0f + e);
}

// B fragment (32x16 K-major per ISA 7.12.2): lane n<16 -> N=n, K={k0..k0+7, k0+16..k0+23};
// lane n>=16 -> N=n-16, K={k0+8..k0+15, k0+24..k0+31}. Caller passes p = &W[row(N)][k0 + 8*half].
static __device__ __forceinline__ v16bf load_b_frag_f32(const float* __restrict__ p) {
    const float4 f0 = ((const float4*)p)[0];
    const float4 f1 = ((const float4*)p)[1];
    const float4 f2 = ((const float4*)p)[4];
    const float4 f3 = ((const float4*)p)[5];
    v16bf b;
    b[0]  = (bf16)f0.x; b[1]  = (bf16)f0.y; b[2]  = (bf16)f0.z; b[3]  = (bf16)f0.w;
    b[4]  = (bf16)f1.x; b[5]  = (bf16)f1.y; b[6]  = (bf16)f1.z; b[7]  = (bf16)f1.w;
    b[8]  = (bf16)f2.x; b[9]  = (bf16)f2.y; b[10] = (bf16)f2.z; b[11] = (bf16)f2.w;
    b[12] = (bf16)f3.x; b[13] = (bf16)f3.y; b[14] = (bf16)f3.z; b[15] = (bf16)f3.w;
    return b;
}

// A fragment (16x32 bf16 per ISA 7.12.2): lane m<16 -> row m, K={k0..k0+7, k0+16..k0+23};
// lane m>=16 -> row m-16, K={k0+8..k0+15, k0+24..k0+31}. p = &row[k0 + 8*half].
static __device__ __forceinline__ v16bf load_a_frag_lds(const bf16* p) {
    v8bf lo = *(const v8bf*)p;
    v8bf hi = *(const v8bf*)(p + 16);
    return __builtin_shufflevector(lo, hi, 0,1,2,3,4,5,6,7,8,9,10,11,12,13,14,15);
}

// TDM gather-mode load (ISA 8.3/8.4/8.7): fetch 16 rows of a [T_TOKENS x HIDDEN]
// fp32 tensor, selected by 16-bit row indices, into LDS at lds_byte_addr.
static __device__ __forceinline__ void tdm_gather_rows16(unsigned lds_byte_addr,
                                                         const float* tensor_base,
                                                         const int* rowidx) {
    unsigned long long ga = (unsigned long long)(uintptr_t)tensor_base;

    v4u g0;
    g0[0] = 0x80000001u;                                    // count=1 | gather_mode=1 (16-bit idx)
    g0[1] = lds_byte_addr;                                  // bits 63:32  lds_addr
    g0[2] = (unsigned)(ga & 0xFFFFFFFFu);                   // bits 95:64  global_addr lo
    g0[3] = (unsigned)((ga >> 32) & 0x01FFFFFFu)            // bits 120:96 global_addr hi
          | 0x80000000u;                                    // bits 127:126 type=2 ("image")

    v8i g1;
    g1[0] = (int)(2u << 16);                                // workgroup_mask=0, data_size=2 (4B)
    g1[1] = (int)((HIDDEN & 0xFFFFu) << 16);                // tensor_dim0[15:0] @ bits 63:48
    g1[2] = (int)((HIDDEN >> 16) |
                  ((T_TOKENS & 0xFFFFu) << 16));            // tensor_dim0 hi | tensor_dim1 lo
    g1[3] = (int)((T_TOKENS >> 16) |
                  ((HIDDEN & 0xFFFFu) << 16));              // tensor_dim1 hi | tile_dim0
    g1[4] = 16;                                             // tile_dim1 = #valid indices
    g1[5] = HIDDEN;                                         // tensor_dim0_stride lo32
    g1[6] = 0;                                              // stride hi | dim1_stride (ignored)
    g1[7] = 0;

    v4i g2, g3;                                             // 16-bit row indices 0..15
    #pragma unroll
    for (int j = 0; j < 4; ++j)
        g2[j] = (rowidx[2 * j] & 0xFFFF) | (rowidx[2 * j + 1] << 16);
    #pragma unroll
    for (int j = 0; j < 4; ++j)
        g3[j] = (rowidx[8 + 2 * j] & 0xFFFF) | (rowidx[8 + 2 * j + 1] << 16);

    const v8i gz = (v8i){0, 0, 0, 0, 0, 0, 0, 0};           // unused trailing group (clang-23 arity)
    __builtin_amdgcn_tensor_load_to_lds(g0, g1, g2, g3, gz, 0);
}

// ---------------- kernel 0: zero output + routing counters ----------------
__global__ __launch_bounds__(256) void moe_zero_kernel(float* __restrict__ out,
                                                       int* __restrict__ counts, int n4) {
    int i = blockIdx.x * blockDim.x + threadIdx.x;
    if (i < n4) ((float4*)out)[i] = make_float4(0.f, 0.f, 0.f, 0.f);
    if (i < NEXPERT) counts[i] = 0;
}

// ---------------- kernel 1: build per-expert token lists ----------------
__global__ __launch_bounds__(256) void moe_route_kernel(const int* __restrict__ topk_idx,
                                                        const float* __restrict__ topk_w,
                                                        int* __restrict__ counts,
                                                        int* __restrict__ toklist,
                                                        float* __restrict__ wlist) {
    int i = blockIdx.x * blockDim.x + threadIdx.x;
    if (i >= T_TOKENS * TOPK) return;
    int e = topk_idx[i];
    int t = i >> 1;                     // token id ([T][K] row-major)
    int slot = atomicAdd(&counts[e], 1);
    toklist[e * MAX_ROWS + slot] = t;
    wlist[e * MAX_ROWS + slot]   = topk_w[i];
}

// ---------------- kernel 2: gathered expert MLP (WMMA bf16 + TDM staging) ----------------
__global__ __launch_bounds__(256)
void moe_mlp_kernel(const float* __restrict__ hidden,
                    const float* __restrict__ gup,     // [E][2*INTER][HIDDEN]
                    const float* __restrict__ dwn,     // [E][HIDDEN][INTER]
                    const int* __restrict__ counts,
                    const int* __restrict__ toklist,
                    const float* __restrict__ wlist,
                    float* __restrict__ out) {
    const int e    = blockIdx.y;
    const int tile = blockIdx.x;
    const int cnt  = counts[e];
    if (tile * MTILE >= cnt) return;

    extern __shared__ char smem[];
    bf16*  Xs    = (bf16*)smem;                              // [MTILE][XS_STRIDE] bf16
    float* Sf    = (float*)(smem + XS_BYTES);                // fp32 TDM landing pad (16 rows)
    bf16*  Hs    = (bf16*)(smem + XS_BYTES);                 // overlaid: live only in main loop
    int*   tok_s = (int*)(smem + XS_BYTES + STAGE_BYTES);    // [MTILE]
    float* wt_s  = (float*)(smem + XS_BYTES + STAGE_BYTES + MTILE * 4);

    const int tid  = threadIdx.x;
    const int wave = tid >> 5;
    const int lane = tid & 31;
    const int half = lane >> 4;
    const int l15  = lane & 15;

    if (tid < MTILE) {
        int r = tile * MTILE + tid;
        bool valid = (r < cnt);
        tok_s[tid] = valid ? toklist[e * MAX_ROWS + r] : 0;  // pad rows gather row 0, weight 0
        wt_s[tid]  = valid ? wlist[e * MAX_ROWS + r]   : 0.0f;
    }
    __syncthreads();

    // ---- stage X tile: TDM row-gather (fp32) -> convert to bf16 ----
    const unsigned stage_lds = (unsigned)(uintptr_t)Sf;
    for (int pass = 0; pass < 2; ++pass) {
        if (wave == 0) {
            tdm_gather_rows16(stage_lds, hidden, tok_s + 16 * pass);
            __builtin_amdgcn_s_wait_tensorcnt(0);
        }
        __syncthreads();
        for (int i = tid; i < 16 * (HIDDEN / 4); i += 256) {
            int row = i >> 8;           // HIDDEN/4 == 256
            int c4  = i & 255;
            float4 v = ((const float4*)Sf)[i];
            v4bf b;
            b[0] = (bf16)v.x; b[1] = (bf16)v.y; b[2] = (bf16)v.z; b[3] = (bf16)v.w;
            *(v4bf*)(Xs + (16 * pass + row) * XS_STRIDE + c4 * 4) = b;
        }
        __syncthreads();
    }

    const float* Wg = gup + (size_t)e * (2 * INTER) * HIDDEN;  // gate rows 0..I-1, up rows I..2I-1
    const float* Wd = dwn + (size_t)e * HIDDEN * INTER;

    v8f acc2[8][2];                     // persistent output accumulators: 8 col-tiles x 2 row-frags
    #pragma unroll
    for (int i = 0; i < 8; ++i) {
        acc2[i][0] = (v8f){0.f,0.f,0.f,0.f,0.f,0.f,0.f,0.f};
        acc2[i][1] = (v8f){0.f,0.f,0.f,0.f,0.f,0.f,0.f,0.f};
    }

    for (int kc = 0; kc < NCHUNK; ++kc) {
        const int c0chunk = kc * CHUNK;

        // ---- GEMM1: gate/up pair-tiles for this chunk (each B frag feeds 2 WMMAs) ----
        #pragma unroll
        for (int ptl = 0; ptl < 4; ++ptl) {
            const int c0 = c0chunk + (wave * 4 + ptl) * 16;    // gate column base
            const float* wg_row = Wg + (size_t)(c0 + l15) * HIDDEN + 8 * half;
            const float* wu_row = Wg + (size_t)(INTER + c0 + l15) * HIDDEN + 8 * half;

            v8f accg0 = (v8f){0.f,0.f,0.f,0.f,0.f,0.f,0.f,0.f};
            v8f accg1 = accg0, accu0 = accg0, accu1 = accg0;

            for (int k0 = 0; k0 < HIDDEN; k0 += 32) {
                __builtin_prefetch(wg_row + k0 + 64, 0, 1);
                v16bf a0 = load_a_frag_lds(Xs + l15 * XS_STRIDE        + k0 + 8 * half);
                v16bf a1 = load_a_frag_lds(Xs + (16 + l15) * XS_STRIDE + k0 + 8 * half);
                v16bf bg = load_b_frag_f32(wg_row + k0);
                v16bf bu = load_b_frag_f32(wu_row + k0);
                accg0 = __builtin_amdgcn_wmma_f32_16x16x32_bf16(false, a0, false, bg, (short)0, accg0, false, false);
                accg1 = __builtin_amdgcn_wmma_f32_16x16x32_bf16(false, a1, false, bg, (short)0, accg1, false, false);
                accu0 = __builtin_amdgcn_wmma_f32_16x16x32_bf16(false, a0, false, bu, (short)0, accu0, false, false);
                accu1 = __builtin_amdgcn_wmma_f32_16x16x32_bf16(false, a1, false, bu, (short)0, accu1, false, false);
            }

            // activation + store h to LDS (C layout: lane<16 -> M=r, lane>=16 -> M=8+r; N=l15)
            const int hcol = (wave * 4 + ptl) * 16 + l15;
            #pragma unroll
            for (int r = 0; r < 8; ++r) {
                float h0 = gelu_tanh(accg0[r]) * accu0[r];
                float h1 = gelu_tanh(accg1[r]) * accu1[r];
                Hs[(8 * half + r) * HS_STRIDE + hcol]      = (bf16)h0;
                Hs[(16 + 8 * half + r) * HS_STRIDE + hcol] = (bf16)h1;
            }
        }
        __syncthreads();

        // ---- GEMM2 partial: consume Hs chunk, accumulate output tile in regs ----
        #pragma unroll
        for (int ntl = 0; ntl < 8; ++ntl) {
            const int ncol = (wave * 8 + ntl) * 16;            // output column base
            const float* wd_row = Wd + (size_t)(ncol + l15) * INTER + c0chunk + 8 * half;
            v8f acc0 = acc2[ntl][0];
            v8f acc1 = acc2[ntl][1];
            for (int k0 = 0; k0 < CHUNK; k0 += 32) {
                __builtin_prefetch(wd_row + k0 + 64, 0, 1);
                v16bf a0 = load_a_frag_lds(Hs + l15 * HS_STRIDE        + k0 + 8 * half);
                v16bf a1 = load_a_frag_lds(Hs + (16 + l15) * HS_STRIDE + k0 + 8 * half);
                v16bf b  = load_b_frag_f32(wd_row + k0);
                acc0 = __builtin_amdgcn_wmma_f32_16x16x32_bf16(false, a0, false, b, (short)0, acc0, false, false);
                acc1 = __builtin_amdgcn_wmma_f32_16x16x32_bf16(false, a1, false, b, (short)0, acc1, false, false);
            }
            acc2[ntl][0] = acc0;
            acc2[ntl][1] = acc1;
        }
        __syncthreads();
    }

    // ---- scale by routing weight, accumulate to output ----
    #pragma unroll
    for (int ntl = 0; ntl < 8; ++ntl) {
        const int ncol = (wave * 8 + ntl) * 16 + l15;
        #pragma unroll
        for (int rt = 0; rt < 2; ++rt) {
            #pragma unroll
            for (int r = 0; r < 8; ++r) {
                const int m = 16 * rt + 8 * half + r;
                float val = wt_s[m] * acc2[ntl][rt][r];
                atomicAdd(&out[(size_t)tok_s[m] * HIDDEN + ncol], val);
            }
        }
    }
}

// ---------------- host launcher ----------------
extern "C" void kernel_launch(void* const* d_in, const int* in_sizes, int n_in,
                              void* d_out, int out_size, void* d_ws, size_t ws_size,
                              hipStream_t stream) {
    const float* hidden   = (const float*)d_in[0];
    const int*   topk_idx = (const int*)d_in[1];
    const float* topk_w   = (const float*)d_in[2];
    const float* gup      = (const float*)d_in[3];
    const float* dwn      = (const float*)d_in[4];
    float* out = (float*)d_out;

    char* ws = (char*)d_ws;
    int*   counts  = (int*)ws;                                          // 8 ints (padded to 64B)
    int*   toklist = (int*)(ws + 64);                                   // [E][MAX_ROWS]
    float* wlist   = (float*)(ws + 64 + (size_t)NEXPERT * MAX_ROWS * 4);// [E][MAX_ROWS]

    const int n4 = T_TOKENS * HIDDEN / 4;
    moe_zero_kernel<<<(n4 + 255) / 256, 256, 0, stream>>>(out, counts, n4);
    moe_route_kernel<<<(T_TOKENS * TOPK + 255) / 256, 256, 0, stream>>>(
        topk_idx, topk_w, counts, toklist, wlist);

    dim3 grid(MAX_TILES, NEXPERT);
    moe_mlp_kernel<<<grid, 256, LDS_BYTES, stream>>>(hidden, gup, dwn, counts, toklist, wlist, out);
}
